// QMixtralDecoderLayer_4475355922940
// MI455X (gfx1250) — matrix operand order repaired and checked by reading, not compiled, not run on hardware
//
#include <hip/hip_runtime.h>

// ---------------- problem dims (match reference) ----------------
constexpr int Bb   = 2;
constexpr int Ss   = 1024;
constexpr int Tt   = Bb * Ss;         // 2048 tokens
constexpr int Dd   = 1024;
constexpr int Hh   = 8;
constexpr int KVHh = 2;
constexpr int HDh  = 128;
constexpr int Ee   = 8;
constexpr int FFNf = 3584;
constexpr float EPSf   = 1e-5f;
constexpr float NEGf   = -1e30f;
constexpr float SCALEf = 0.08838834764831845f;   // 1/sqrt(128)

typedef __attribute__((ext_vector_type(16))) _Float16 v16h;
typedef __attribute__((ext_vector_type(8)))  _Float16 v8h;
typedef __attribute__((ext_vector_type(8)))  float    v8f;
typedef __attribute__((ext_vector_type(4)))  int      v4i;

// CDNA5 async global->LDS path (ASYNCcnt-tracked), guarded so the kernel
// still builds on toolchains without the builtin.
#if defined(__has_builtin)
#if __has_builtin(__builtin_amdgcn_global_load_async_to_lds_b128)
#define HAVE_ASYNC_LDS 1
#endif
#if __has_builtin(__builtin_amdgcn_s_wait_asynccnt)
#define HAVE_WAIT_ASYNC 1
#endif
#endif

#ifdef HAVE_ASYNC_LDS
// builtin prototype (from clang diagnostic): param0 = global int4*, param1 = LDS int4*
typedef __attribute__((address_space(1))) v4i* gas_ptr;
typedef __attribute__((address_space(3))) v4i* las_ptr;
#endif

__device__ __forceinline__ void wait_async_zero() {
#ifdef HAVE_ASYNC_LDS
#ifdef HAVE_WAIT_ASYNC
  __builtin_amdgcn_s_wait_asynccnt(0);
#else
  asm volatile("s_wait_asynccnt 0x0" ::: "memory");
#endif
#endif
}

// ---------------- WMMA GEMM core:  C = A(MxK,f16) * B(NxK,f16)^T ----------------
// Block tile 128x128, K-tile 64, 256 threads = 8 waves in a 2(M) x 4(N) grid,
// each wave computes 64x32 via a 4x2 array of 16x16 WMMA fragments.
constexpr int BM = 128, BN = 128, BK = 64;

enum { M_STORE_F16 = 0, M_SCORES = 1, M_ADD_RES = 2, M_MOE_DOWN = 3 };

template <int MODE>
__device__ __forceinline__ void gemm_core(
    const _Float16* __restrict__ A, int lda,
    const _Float16* __restrict__ B, int ldb,
    void* __restrict__ Cout, int ldc, int K,
    const float* __restrict__ res,      // M_ADD_RES: residual, same layout as C
    const int*   __restrict__ rowMap,   // optional token gather for A rows / scatter
    const float* __restrict__ rowW)     // M_MOE_DOWN: per-row routing weight
{
  __shared__ _Float16 As[BM][BK + 8];
  __shared__ _Float16 Bs[BN][BK + 8];

  const int tid  = threadIdx.x;
  const int lane = tid & 31;
  const int wave = tid >> 5;
  const int wm   = wave >> 2;          // 0..1
  const int wn   = wave & 3;           // 0..3
  const int bm   = blockIdx.y * BM;
  const int bn   = blockIdx.x * BN;

  v8f acc[4][2];
#pragma unroll
  for (int mi = 0; mi < 4; mi++)
#pragma unroll
    for (int ni = 0; ni < 2; ni++)
#pragma unroll
      for (int j = 0; j < 8; j++) acc[mi][ni][j] = 0.0f;

  for (int k0 = 0; k0 < K; k0 += BK) {
    // stage A and B tiles in LDS: 128x64 halfs each, 4x 16B chunks per thread
#ifdef HAVE_ASYNC_LDS
#pragma unroll
    for (int i = 0; i < 4; i++) {
      int cid = tid + i * 256;
      int r   = cid >> 3;
      int c8  = (cid & 7) * 8;
      int ga  = rowMap ? rowMap[bm + r] : (bm + r);
      __builtin_amdgcn_global_load_async_to_lds_b128(
          (gas_ptr)&A[(size_t)ga * lda + k0 + c8], (las_ptr)&As[r][c8], 0, 0);
      __builtin_amdgcn_global_load_async_to_lds_b128(
          (gas_ptr)&B[(size_t)(bn + r) * ldb + k0 + c8], (las_ptr)&Bs[r][c8], 0, 0);
    }
    wait_async_zero();
#else
#pragma unroll
    for (int i = 0; i < 4; i++) {
      int cid = tid + i * 256;
      int r   = cid >> 3;
      int c8  = (cid & 7) * 8;
      int ga  = rowMap ? rowMap[bm + r] : (bm + r);
      *(v8h*)&As[r][c8] = *(const v8h*)&A[(size_t)ga * lda + k0 + c8];
      *(v8h*)&Bs[r][c8] = *(const v8h*)&B[(size_t)(bn + r) * ldb + k0 + c8];
    }
#endif
    if (k0 + BK < K) {   // hint: pull next weight K-tile toward L2 (global_prefetch_b8)
      __builtin_prefetch(&B[(size_t)(bn + (tid >> 1)) * ldb + k0 + BK + (tid & 1) * 32], 0, 1);
    }
    __syncthreads();

#pragma unroll
    for (int ks = 0; ks < BK / 32; ks++) {
      v16h af[4], bf[2];
#pragma unroll
      for (int mi = 0; mi < 4; mi++) {
        // A-frag (16x32): lane l -> row l%16 ; K halves {0..7,16..23} / {8..15,24..31}
        const _Float16* p = &As[wm * 64 + mi * 16 + (lane & 15)][ks * 32 + (lane >> 4) * 8];
        v8h lo = *(const v8h*)p;
        v8h hi = *(const v8h*)(p + 16);
#pragma unroll
        for (int j = 0; j < 8; j++) { af[mi][j] = lo[j]; af[mi][j + 8] = hi[j]; }
      }
#pragma unroll
      for (int ni = 0; ni < 2; ni++) {
        // B-frag (32x16): lane l -> col l%16 ; K halves {0..7,16..23} / {8..15,24..31}
        const _Float16* p = &Bs[wn * 32 + ni * 16 + (lane & 15)][ks * 32 + (lane >> 4) * 8];
        v8h lo = *(const v8h*)p;
        v8h hi = *(const v8h*)(p + 16);
#pragma unroll
        for (int j = 0; j < 8; j++) { bf[ni][j] = lo[j]; bf[ni][j + 8] = hi[j]; }
      }
#pragma unroll
      for (int mi = 0; mi < 4; mi++)
#pragma unroll
        for (int ni = 0; ni < 2; ni++)
          acc[mi][ni] = __builtin_amdgcn_wmma_f32_16x16x32_f16(
              false, af[mi], false, bf[ni], (short)0, acc[mi][ni], false, false);
    }
    __syncthreads();
  }

  // epilogue (C layout: lane -> col lane%16, VGPR r -> row 8*(lane/16)+r)
#pragma unroll
  for (int mi = 0; mi < 4; mi++)
#pragma unroll
    for (int ni = 0; ni < 2; ni++) {
      int m0 = bm + wm * 64 + mi * 16 + (lane >> 4) * 8;
      int n  = bn + wn * 32 + ni * 16 + (lane & 15);
#pragma unroll
      for (int r = 0; r < 8; r++) {
        int   m = m0 + r;
        float v = acc[mi][ni][r];
        if (MODE == M_STORE_F16) {
          ((_Float16*)Cout)[(size_t)m * ldc + n] = (_Float16)v;
        } else if (MODE == M_SCORES) {
          v *= SCALEf;
          if (n > m) v = NEGf;                       // causal mask
          ((float*)Cout)[(size_t)m * ldc + n] = v;
        } else if (MODE == M_ADD_RES) {
          ((float*)Cout)[(size_t)m * ldc + n] = v + res[(size_t)m * ldc + n];
        } else if (MODE == M_MOE_DOWN) {
          int tok = rowMap[m];
          ((float*)Cout)[(size_t)tok * ldc + n] += rowW[m] * v;  // unique (tok,n) per launch
        }
      }
    }
}

// ---------------- GEMM wrapper kernels ----------------
__global__ void __launch_bounds__(256)
k_gemm_f16(const _Float16* A, int lda, const _Float16* B, int ldb,
           _Float16* C, int ldc, int K, const int* rowMap, const int* cnt) {
  if (cnt && (int)(blockIdx.y * BM) >= *cnt) return;   // gathered-expert early exit
  gemm_core<M_STORE_F16>(A, lda, B, ldb, C, ldc, K, nullptr, rowMap, nullptr);
}

__global__ void __launch_bounds__(256)
k_gemm_scores(const _Float16* q, const _Float16* kk, float* sc) {
  int z = blockIdx.z, b = z / Hh, h = z % Hh, kv = h / (Hh / KVHh);
  const _Float16* A = q  + (size_t)b * Ss * (Hh * HDh)   + (size_t)h  * HDh;
  const _Float16* B = kk + (size_t)b * Ss * (KVHh * HDh) + (size_t)kv * HDh;
  float* C = sc + (size_t)z * Ss * Ss;
  gemm_core<M_SCORES>(A, Hh * HDh, B, KVHh * HDh, C, Ss, HDh, nullptr, nullptr, nullptr);
}

__global__ void __launch_bounds__(256)
k_gemm_pv(const _Float16* pr, const _Float16* vT, _Float16* ob) {
  int z = blockIdx.z, b = z / Hh, h = z % Hh, kv = h / (Hh / KVHh);
  const _Float16* A = pr + (size_t)z * Ss * Ss;
  const _Float16* B = vT + (size_t)(b * KVHh + kv) * HDh * Ss;
  _Float16* C = ob + (size_t)b * Ss * (Hh * HDh) + (size_t)h * HDh;
  gemm_core<M_STORE_F16>(A, Ss, B, Ss, C, Hh * HDh, Ss, nullptr, nullptr, nullptr);
}

__global__ void __launch_bounds__(256)
k_gemm_attnout(const _Float16* ob, const _Float16* wo, const float* resid, float* hb) {
  gemm_core<M_ADD_RES>(ob, Hh * HDh, wo, Dd, hb, Dd, Hh * HDh, resid, nullptr, nullptr);
}

__global__ void __launch_bounds__(256)
k_gemm_down(const _Float16* act, const _Float16* w2e, float* outp,
            const int* gidx, const float* gw, const int* cnt) {
  if ((int)(blockIdx.y * BM) >= *cnt) return;
  gemm_core<M_MOE_DOWN>(act, FFNf, w2e, FFNf, outp, Dd, FFNf, nullptr, gidx, gw);
}

// ---------------- elementwise / reduction kernels ----------------
__global__ void __launch_bounds__(256)
k_rmsnorm(const float* __restrict__ x, const float* __restrict__ g,
          _Float16* __restrict__ o16, float* __restrict__ o32) {
  const int t = blockIdx.x;
  const float* xr = x + (size_t)t * Dd;
  float v[4], ss = 0.f;
#pragma unroll
  for (int i = 0; i < 4; i++) { v[i] = xr[threadIdx.x + i * 256]; ss += v[i] * v[i]; }
  for (int o = 16; o > 0; o >>= 1) ss += __shfl_xor(ss, o, 32);
  __shared__ float red[8];
  if ((threadIdx.x & 31) == 0) red[threadIdx.x >> 5] = ss;
  __syncthreads();
  float tot = 0.f;
#pragma unroll
  for (int i = 0; i < 8; i++) tot += red[i];
  float r = rsqrtf(tot / (float)Dd + EPSf);
#pragma unroll
  for (int i = 0; i < 4; i++) {
    int c = threadIdx.x + i * 256;
    float y = v[i] * r * g[c];
    o16[(size_t)t * Dd + c] = (_Float16)y;
    if (o32) o32[(size_t)t * Dd + c] = y;
  }
}

// int4 asymmetric group fake-quant (group=128 contiguous), f32 -> f16
__global__ void __launch_bounds__(128)
k_quantw(const float* __restrict__ w, _Float16* __restrict__ o) {
  const size_t g = blockIdx.x;
  const int t = threadIdx.x;
  float v = w[g * 128 + t];
  float mn = v, mx = v;
  for (int s = 16; s > 0; s >>= 1) {
    mn = fminf(mn, __shfl_xor(mn, s, 32));
    mx = fmaxf(mx, __shfl_xor(mx, s, 32));
  }
  __shared__ float smn[4], smx[4];
  if ((t & 31) == 0) { smn[t >> 5] = mn; smx[t >> 5] = mx; }
  __syncthreads();
  mn = fminf(fminf(smn[0], smn[1]), fminf(smn[2], smn[3]));
  mx = fmaxf(fmaxf(smx[0], smx[1]), fmaxf(smx[2], smx[3]));
  float scale = fmaxf(mx - mn, 1e-5f) * (1.f / 15.f);
  float base  = fminf(fmaxf(rintf(-mn / scale), 0.f), 15.f);
  float q     = fminf(fmaxf(rintf(v / scale) + base, 0.f), 15.f);
  o[g * 128 + t] = (_Float16)((q - base) * scale);
}

// same fake-quant, in-place on f16 activations (k before RoPE, v)
__global__ void __launch_bounds__(128)
k_quanta(_Float16* __restrict__ x) {
  const size_t g = blockIdx.x;
  const int t = threadIdx.x;
  float v = (float)x[g * 128 + t];
  float mn = v, mx = v;
  for (int s = 16; s > 0; s >>= 1) {
    mn = fminf(mn, __shfl_xor(mn, s, 32));
    mx = fmaxf(mx, __shfl_xor(mx, s, 32));
  }
  __shared__ float smn[4], smx[4];
  if ((t & 31) == 0) { smn[t >> 5] = mn; smx[t >> 5] = mx; }
  __syncthreads();
  mn = fminf(fminf(smn[0], smn[1]), fminf(smn[2], smn[3]));
  mx = fmaxf(fmaxf(smx[0], smx[1]), fmaxf(smx[2], smx[3]));
  float scale = fmaxf(mx - mn, 1e-5f) * (1.f / 15.f);
  float base  = fminf(fmaxf(rintf(-mn / scale), 0.f), 15.f);
  float q     = fminf(fmaxf(rintf(v / scale) + base, 0.f), 15.f);
  x[g * 128 + t] = (_Float16)((q - base) * scale);
}

__global__ void __launch_bounds__(256)
k_rope(_Float16* __restrict__ x, int NH) {
  int idx = blockIdx.x * 256 + threadIdx.x;
  int total = Bb * Ss * NH * (HDh / 2);
  if (idx >= total) return;
  int i = idx & 63;
  int rest = idx >> 6;                 // (b*S+s)*NH + h
  int s = (rest / NH) % Ss;
  float inv = __powf(1.0e6f, -(float)(2 * i) / (float)HDh);
  float fr = (float)s * inv;
  float sn, c;
  __sincosf(fr, &sn, &c);
  size_t b0 = (size_t)rest * HDh;
  float x0 = (float)x[b0 + i], x1 = (float)x[b0 + i + 64];
  x[b0 + i]      = (_Float16)(x0 * c - x1 * sn);
  x[b0 + i + 64] = (_Float16)(x1 * c + x0 * sn);
}

__global__ void
k_transpose_v(const _Float16* __restrict__ v, _Float16* __restrict__ vT) {
  __shared__ _Float16 tile[32][33];
  int bkv = blockIdx.z, b = bkv / KVHh, kv = bkv % KVHh;
  int s0 = blockIdx.x * 32, d0 = blockIdx.y * 32;
  int tx = threadIdx.x, ty = threadIdx.y;
#pragma unroll
  for (int i = 0; i < 32; i += 8) {
    int s = s0 + ty + i, d = d0 + tx;
    tile[ty + i][tx] = v[((size_t)(b * Ss + s) * KVHh + kv) * HDh + d];
  }
  __syncthreads();
#pragma unroll
  for (int i = 0; i < 32; i += 8) {
    int d = d0 + ty + i, s = s0 + tx;
    vT[((size_t)(b * KVHh + kv) * HDh + d) * Ss + s] = tile[tx][ty + i];
  }
}

__global__ void __launch_bounds__(256)
k_softmax_row(const float* __restrict__ sc, _Float16* __restrict__ pr) {
  const size_t row = blockIdx.x;
  const float* s = sc + row * Ss;
  float v[4], mx = -3.4e38f;
#pragma unroll
  for (int i = 0; i < 4; i++) { v[i] = s[threadIdx.x + i * 256]; mx = fmaxf(mx, v[i]); }
  for (int o = 16; o > 0; o >>= 1) mx = fmaxf(mx, __shfl_xor(mx, o, 32));
  __shared__ float red[8];
  if ((threadIdx.x & 31) == 0) red[threadIdx.x >> 5] = mx;
  __syncthreads();
  mx = red[0];
#pragma unroll
  for (int i = 1; i < 8; i++) mx = fmaxf(mx, red[i]);
  float sum = 0.f;
#pragma unroll
  for (int i = 0; i < 4; i++) { v[i] = __expf(v[i] - mx); sum += v[i]; }
  for (int o = 16; o > 0; o >>= 1) sum += __shfl_xor(sum, o, 32);
  __syncthreads();
  if ((threadIdx.x & 31) == 0) red[threadIdx.x >> 5] = sum;
  __syncthreads();
  sum = 0.f;
#pragma unroll
  for (int i = 0; i < 8; i++) sum += red[i];
  float inv = 1.f / sum;
#pragma unroll
  for (int i = 0; i < 4; i++)
    pr[row * Ss + threadIdx.x + i * 256] = (_Float16)(v[i] * inv);
}

__global__ void k_copy4(const float* __restrict__ a, float* __restrict__ b, int n4) {
  int i = blockIdx.x * 256 + threadIdx.x;
  if (i < n4) ((float4*)b)[i] = ((const float4*)a)[i];
}

__global__ void k_init_gather(int* cnt, int* gidx, float* gw) {
  int i = blockIdx.x * 256 + threadIdx.x;
  if (i < Ee) cnt[i] = 0;
  if (i < Ee * Tt) { gidx[i] = 0; gw[i] = 0.f; }
}

// router: full-precision logits, softmax over 8, top-2, normalized weights,
// atomic gather into per-expert compacted token lists.
__global__ void __launch_bounds__(256)
k_router(const float* __restrict__ hn, const float* __restrict__ wg,
         int* cnt, int* gidx, float* gw) {
  int t = blockIdx.x;
  int e = threadIdx.x >> 5, lane = threadIdx.x & 31;
  const float* x = hn + (size_t)t * Dd;
  const float* w = wg + (size_t)e * Dd;
  float s = 0.f;
  for (int i = lane; i < Dd; i += 32) s += x[i] * w[i];
  for (int o = 16; o > 0; o >>= 1) s += __shfl_xor(s, o, 32);
  __shared__ float logit[Ee];
  if (lane == 0) logit[e] = s;
  __syncthreads();
  if (threadIdx.x == 0) {
    float mx = logit[0];
    for (int i = 1; i < Ee; i++) mx = fmaxf(mx, logit[i]);
    float p[Ee], sum = 0.f;
    for (int i = 0; i < Ee; i++) { p[i] = __expf(logit[i] - mx); sum += p[i]; }
    for (int i = 0; i < Ee; i++) p[i] /= sum;
    int e0 = 0;
    for (int i = 1; i < Ee; i++) if (p[i] > p[e0]) e0 = i;
    int e1 = -1;
    for (int i = 0; i < Ee; i++) { if (i == e0) continue; if (e1 < 0 || p[i] > p[e1]) e1 = i; }
    float w0 = p[e0], w1 = p[e1], wsum = w0 + w1;
    w0 /= wsum; w1 /= wsum;
    int p0 = atomicAdd(&cnt[e0], 1); gidx[e0 * Tt + p0] = t; gw[e0 * Tt + p0] = w0;
    int p1 = atomicAdd(&cnt[e1], 1); gidx[e1 * Tt + p1] = t; gw[e1 * Tt + p1] = w1;
  }
}

__global__ void k_silu_mul(const _Float16* __restrict__ g, const _Float16* __restrict__ u,
                           _Float16* __restrict__ a, int n) {
  int i = blockIdx.x * 256 + threadIdx.x;
  if (i < n) {
    float gv = (float)g[i], uv = (float)u[i];
    a[i] = (_Float16)(gv / (1.f + __expf(-gv)) * uv);
  }
}

// ---------------- launcher ----------------
// Workspace requirement: ~140 MB (MoE buffers overlay the dead score/prob arena).
extern "C" void kernel_launch(void* const* d_in, const int* in_sizes, int n_in,
                              void* d_out, int out_size, void* d_ws, size_t ws_size,
                              hipStream_t stream) {
  (void)in_sizes; (void)n_in; (void)out_size; (void)ws_size;
  const float* hidden = (const float*)d_in[0];
  const float* w_q    = (const float*)d_in[1];
  const float* w_k    = (const float*)d_in[2];
  const float* w_v    = (const float*)d_in[3];
  const float* w_o    = (const float*)d_in[4];
  const float* g1     = (const float*)d_in[5];
  const float* g2     = (const float*)d_in[6];
  const float* w_gate = (const float*)d_in[7];
  const float* w1     = (const float*)d_in[8];
  const float* w3     = (const float*)d_in[9];
  const float* w2     = (const float*)d_in[10];
  float* outp = (float*)d_out;

  char* ws = (char*)d_ws;
  size_t off = 0;
  auto alloc = [&](size_t bytes) -> char* {
    off = (off + 255) & ~(size_t)255;
    char* p = ws + off;
    off += bytes;
    return p;
  };

  _Float16* xn16  = (_Float16*)alloc((size_t)Tt * Dd * 2);
  _Float16* wq16  = (_Float16*)alloc((size_t)Dd * Dd * 2);
  _Float16* wk16  = (_Float16*)alloc((size_t)KVHh * HDh * Dd * 2);
  _Float16* wv16  = (_Float16*)alloc((size_t)KVHh * HDh * Dd * 2);
  _Float16* wo16  = (_Float16*)alloc((size_t)Dd * Dd * 2);
  _Float16* qbuf  = (_Float16*)alloc((size_t)Tt * Hh * HDh * 2);
  _Float16* kbuf  = (_Float16*)alloc((size_t)Tt * KVHh * HDh * 2);
  _Float16* vbuf  = (_Float16*)alloc((size_t)Tt * KVHh * HDh * 2);
  _Float16* vTb   = (_Float16*)alloc((size_t)Bb * KVHh * HDh * Ss * 2);
  float*    hbuf  = (float*)alloc((size_t)Tt * Dd * 4);
  _Float16* hn16  = (_Float16*)alloc((size_t)Tt * Dd * 2);
  float*    hnf32 = (float*)alloc((size_t)Tt * Dd * 4);
  int*      cnt   = (int*)alloc(Ee * 4);
  int*      gidx  = (int*)alloc((size_t)Ee * Tt * 4);
  float*    gw    = (float*)alloc((size_t)Ee * Tt * 4);
  size_t arenaOff = off;  // everything below is dead by the time MoE runs
  float*    scores = (float*)alloc((size_t)Bb * Hh * Ss * Ss * 4);
  _Float16* probs  = (_Float16*)alloc((size_t)Bb * Hh * Ss * Ss * 2);
  _Float16* obuf   = (_Float16*)alloc((size_t)Tt * Hh * HDh * 2);
  // MoE overlay on the attention arena (66 MB < 100 MB)
  off = arenaOff;
  _Float16* w1e  = (_Float16*)alloc((size_t)FFNf * Dd * 2);
  _Float16* w3e  = (_Float16*)alloc((size_t)FFNf * Dd * 2);
  _Float16* w2e  = (_Float16*)alloc((size_t)Dd * FFNf * 2);
  _Float16* gbuf = (_Float16*)alloc((size_t)Tt * FFNf * 2);
  _Float16* ubuf = (_Float16*)alloc((size_t)Tt * FFNf * 2);
  _Float16* act  = (_Float16*)alloc((size_t)Tt * FFNf * 2);

  // ---- attention block ----
  k_rmsnorm<<<Tt, 256, 0, stream>>>(hidden, g1, xn16, nullptr);
  k_quantw<<<(Dd * Dd) / 128, 128, 0, stream>>>(w_q, wq16);
  k_quantw<<<(KVHh * HDh * Dd) / 128, 128, 0, stream>>>(w_k, wk16);
  k_quantw<<<(KVHh * HDh * Dd) / 128, 128, 0, stream>>>(w_v, wv16);
  k_quantw<<<(Dd * Dd) / 128, 128, 0, stream>>>(w_o, wo16);

  k_gemm_f16<<<dim3(Dd / BN, Tt / BM), 256, 0, stream>>>(
      xn16, Dd, wq16, Dd, qbuf, Dd, Dd, nullptr, nullptr);
  k_gemm_f16<<<dim3((KVHh * HDh) / BN, Tt / BM), 256, 0, stream>>>(
      xn16, Dd, wk16, Dd, kbuf, KVHh * HDh, Dd, nullptr, nullptr);
  k_gemm_f16<<<dim3((KVHh * HDh) / BN, Tt / BM), 256, 0, stream>>>(
      xn16, Dd, wv16, Dd, vbuf, KVHh * HDh, Dd, nullptr, nullptr);

  k_quanta<<<(Tt * KVHh * HDh) / 128, 128, 0, stream>>>(kbuf);  // quantize K before RoPE
  k_rope<<<(Tt * Hh * (HDh / 2)) / 256, 256, 0, stream>>>(qbuf, Hh);
  k_rope<<<(Tt * KVHh * (HDh / 2)) / 256, 256, 0, stream>>>(kbuf, KVHh);
  k_quanta<<<(Tt * KVHh * HDh) / 128, 128, 0, stream>>>(vbuf);  // quantize V
  k_transpose_v<<<dim3(Ss / 32, HDh / 32, Bb * KVHh), dim3(32, 8), 0, stream>>>(vbuf, vTb);

  k_gemm_scores<<<dim3(Ss / BN, Ss / BM, Bb * Hh), 256, 0, stream>>>(qbuf, kbuf, scores);
  k_softmax_row<<<Bb * Hh * Ss, 256, 0, stream>>>(scores, probs);
  k_gemm_pv<<<dim3(HDh / BN, Ss / BM, Bb * Hh), 256, 0, stream>>>(probs, vTb, obuf);
  k_gemm_attnout<<<dim3(Dd / BN, Tt / BM), 256, 0, stream>>>(obuf, wo16, hidden, hbuf);
  k_copy4<<<(Tt * Dd / 4 + 255) / 256, 256, 0, stream>>>(hbuf, outp, Tt * Dd / 4);

  // ---- MoE block ----
  k_rmsnorm<<<Tt, 256, 0, stream>>>(hbuf, g2, hn16, hnf32);
  k_init_gather<<<(Ee * Tt + 255) / 256, 256, 0, stream>>>(cnt, gidx, gw);
  k_router<<<Tt, 256, 0, stream>>>(hnf32, w_gate, cnt, gidx, gw);

  for (int e = 0; e < Ee; e++) {
    k_quantw<<<(FFNf * Dd) / 128, 128, 0, stream>>>(w1 + (size_t)e * FFNf * Dd, w1e);
    k_quantw<<<(FFNf * Dd) / 128, 128, 0, stream>>>(w3 + (size_t)e * FFNf * Dd, w3e);
    k_quantw<<<(Dd * FFNf) / 128, 128, 0, stream>>>(w2 + (size_t)e * Dd * FFNf, w2e);
    k_gemm_f16<<<dim3(FFNf / BN, Tt / BM), 256, 0, stream>>>(
        hn16, Dd, w1e, Dd, gbuf, FFNf, Dd, gidx + e * Tt, cnt + e);
    k_gemm_f16<<<dim3(FFNf / BN, Tt / BM), 256, 0, stream>>>(
        hn16, Dd, w3e, Dd, ubuf, FFNf, Dd, gidx + e * Tt, cnt + e);
    k_silu_mul<<<(Tt * FFNf + 255) / 256, 256, 0, stream>>>(gbuf, ubuf, act, Tt * FFNf);
    k_gemm_down<<<dim3(Dd / BN, Tt / BM), 256, 0, stream>>>(
        act, w2e, outp, gidx + e * Tt, gw + e * Tt, cnt + e);
  }
}
